// SchNet_9216999817564
// MI455X (gfx1250) — matrix operand order repaired and checked by reading, not compiled, not run on hardware
//
#include <hip/hip_runtime.h>
#include <hip/hip_bf16.h>
#include <math.h>

#define NNODES 50000
#define NEDGES 1600000
#define DF 64
#define NBASIS 25
#define NLAYERS 3
#define CUTOFF_R 5.0f

typedef _Float16 v16h __attribute__((ext_vector_type(16)));
typedef _Float16 h8   __attribute__((ext_vector_type(8)));
typedef float    v8f  __attribute__((ext_vector_type(8)));
typedef float    f4   __attribute__((ext_vector_type(4)));

// ---------------------------------------------------------------- WMMA core
__device__ __forceinline__ v8f wmma16(v16h a, v16h b, v8f c) {
  // D = A(16x32 f16) * B(32x16 f16) + C(16x16 f32)
  return __builtin_amdgcn_wmma_f32_16x16x32_f16(false, a, false, b, (short)0, c,
                                                false, false);
}

// A-fragment (16x32 f16) built from a global f32 row (row stride 64 floats).
// Lane L: M = L&15, g = L>>4.  halves 0..7 <- K = g*8..g*8+7,
// halves 8..15 <- K = 16+g*8 .. 16+g*8+7 (relative to chunk base).
__device__ __forceinline__ v16h load_a_f32(const float* __restrict__ row, int g) {
  f4 p0 = *(const f4*)(row + g * 8);
  f4 p1 = *(const f4*)(row + g * 8 + 4);
  f4 p2 = *(const f4*)(row + 16 + g * 8);
  f4 p3 = *(const f4*)(row + 16 + g * 8 + 4);
  v16h a;
#pragma unroll
  for (int i = 0; i < 4; ++i) {
    a[i]      = (_Float16)p0[i];
    a[4 + i]  = (_Float16)p1[i];
    a[8 + i]  = (_Float16)p2[i];
    a[12 + i] = (_Float16)p3[i];
  }
  return a;
}

// A-fragment from an f16 LDS row (two contiguous 16B runs).
__device__ __forceinline__ v16h load_a_lds(const _Float16* row, int g) {
  h8 lo = *(const h8*)(row + g * 8);
  h8 hi = *(const h8*)(row + 16 + g * 8);
  v16h a;
#pragma unroll
  for (int i = 0; i < 8; ++i) { a[i] = lo[i]; a[8 + i] = hi[i]; }
  return a;
}

// B-fragment from transposed f16 weights WT[N][Kp] (n-major, Kp = padded K).
// Lane L: N = 16*t + (L&15); halves 0..7 <- K = c*32 + (L>>4)*8 + h,
// halves 8..15 <- K = c*32 + 16 + (L>>4)*8 + (h-8).
__device__ __forceinline__ v16h load_b(const _Float16* __restrict__ WT, int Kp,
                                       int t, int c, int lane) {
  const _Float16* p = WT + (size_t)(t * 16 + (lane & 15)) * Kp + c * 32 +
                      (lane >> 4) * 8;
  h8 lo = *(const h8*)p;
  h8 hi = *(const h8*)(p + 16);
  v16h b;
#pragma unroll
  for (int i = 0; i < 8; ++i) { b[i] = lo[i]; b[8 + i] = hi[i]; }
  return b;
}

// swish = x * sigmoid(x); use v_rcp_f32 instead of the IEEE divide sequence
// (operands are already f16-quantized in the WMMA pipeline, ~1ulp rcp is fine).
__device__ __forceinline__ float swishf(float v) {
  return v * __builtin_amdgcn_rcpf(1.0f + __expf(-v));
}

__device__ __forceinline__ void atomicMaxF(float* addr, float val) {
  int* ai = (int*)addr;
  int old = __float_as_int(*addr);
  while (val > __int_as_float(old)) {
    int assumed = old;
    old = atomicCAS(ai, assumed, __float_as_int(val));
    if (old == assumed) break;
  }
}

// --------------------------------------------------------- node GEMM (WMMA)
// out[16 rows x 64] per wave: out = [swish](in @ WT^T + bias) [+ resid]
// Optional per-feature block-reduced sum into redbuf[64].
template <bool BIAS, bool SWISH, bool RESID, bool REDUCE>
__global__ void node_gemm_kernel(const float* __restrict__ in,
                                 const _Float16* __restrict__ WT,
                                 const float* __restrict__ bias,
                                 const float* __restrict__ resid,
                                 float* __restrict__ out,
                                 float* __restrict__ redbuf, int nrows) {
  __shared__ float bsum[64];
  const int lane = threadIdx.x & 31;
  const int wave = threadIdx.x >> 5;
  const int tile = blockIdx.x * (blockDim.x >> 5) + wave;
  const int m0 = tile * 16;
  const bool active = (m0 < nrows);
  if (REDUCE) {
    if (threadIdx.x < 64) bsum[threadIdx.x] = 0.f;
    __syncthreads();
  }
  if (active) {
    const int l15 = lane & 15, g = lane >> 4;
    v8f acc[4];
#pragma unroll
    for (int t = 0; t < 4; ++t) {
      float b0 = BIAS ? bias[t * 16 + l15] : 0.f;
#pragma unroll
      for (int r = 0; r < 8; ++r) acc[t][r] = b0;
    }
    const float* arow = in + (size_t)(m0 + l15) * DF;
    v16h a0 = load_a_f32(arow, g);
    v16h a1 = load_a_f32(arow + 32, g);
#pragma unroll
    for (int t = 0; t < 4; ++t) {
      acc[t] = wmma16(a0, load_b(WT, 64, t, 0, lane), acc[t]);
      acc[t] = wmma16(a1, load_b(WT, 64, t, 1, lane), acc[t]);
    }
#pragma unroll
    for (int t = 0; t < 4; ++t) {
      const int f = t * 16 + l15;
      float lsum = 0.f;
#pragma unroll
      for (int r = 0; r < 8; ++r) {
        const int row = m0 + g * 8 + r;  // C layout: M = g*8 + r
        float v = acc[t][r];
        if (SWISH) v = swishf(v);
        if (RESID) v += resid[(size_t)row * DF + f];
        out[(size_t)row * DF + f] = v;
        lsum += v;
      }
      if (REDUCE) atomicAdd(&bsum[f], lsum);
    }
  }
  if (REDUCE) {
    __syncthreads();
    if (threadIdx.x < 64) atomicAdd(&redbuf[threadIdx.x], bsum[threadIdx.x]);
  }
}

// ------------------------------------------------------------- edge kernel
// Per wave: 16-edge tile. fij(16x32, RBF) -> WMMA -> swish -> WMMA ->
// *cutoff -> gather x1[src] -> atomic scatter agg[dst].
__global__ void edge_kernel(const float* __restrict__ rij,
                            const int* __restrict__ src,
                            const int* __restrict__ dst,
                            const float* __restrict__ x1,
                            const _Float16* __restrict__ Wf1T,
                            const float* __restrict__ bf1,
                            const _Float16* __restrict__ Wf2T,
                            const float* __restrict__ bf2,
                            float* __restrict__ agg, int nedges) {
  __shared__ __align__(32) _Float16 Af[4][16 * 32];
  __shared__ __align__(32) _Float16 Ah[4][16 * 64];
  __shared__ float Rr[4][16], Ct[4][16];
  __shared__ int Se[4][16], De[4][16];

  const int lane = threadIdx.x & 31;
  const int w = threadIdx.x >> 5;
  const int tile = blockIdx.x * 4 + w;
  const int e0 = tile * 16;
  const bool active = (e0 < nedges);
  const int l15 = lane & 15, g = lane >> 4;

  if (active && lane < 16) {
    const int e = e0 + lane;
    const float r = rij[e];
    Rr[w][lane] = r;
    Ct[w][lane] =
        (r < CUTOFF_R) ? 0.5f * (__cosf(r * 0.62831853072f) + 1.0f) : 0.0f;
    Se[w][lane] = src[e];
    De[w][lane] = dst[e];
  }
  __syncthreads();

  if (active) {  // build padded fij tile (k>=25 -> 0, matching zero W rows)
    const float width = CUTOFF_R / 24.0f;
    const float coeff = -0.5f / (width * width);
#pragma unroll
    for (int i = 0; i < 16; ++i) {
      const int idx = lane + 32 * i;
      const int ed = idx >> 5, k = idx & 31;
      float v = 0.f;
      if (k < NBASIS) {
        const float d = Rr[w][ed] - (float)k * width;
        v = __expf(coeff * d * d);
      }
      Af[w][ed * 32 + k] = (_Float16)v;
    }
  }
  __syncthreads();

  if (active) {  // GEMM1: H = swish(fij @ Wf1 + bf1)
    v8f hacc[4];
#pragma unroll
    for (int t = 0; t < 4; ++t) {
      float b0 = bf1[t * 16 + l15];
#pragma unroll
      for (int r = 0; r < 8; ++r) hacc[t][r] = b0;
    }
    v16h a = load_a_lds(&Af[w][l15 * 32], g);
#pragma unroll
    for (int t = 0; t < 4; ++t)
      hacc[t] = wmma16(a, load_b(Wf1T, 32, t, 0, lane), hacc[t]);
#pragma unroll
    for (int t = 0; t < 4; ++t) {
      const int f = t * 16 + l15;
#pragma unroll
      for (int r = 0; r < 8; ++r)
        Ah[w][(g * 8 + r) * 64 + f] = (_Float16)swishf(hacc[t][r]);
    }
  }
  __syncthreads();

  if (active) {  // GEMM2 + cutoff + gather/scatter
    v8f acc[4];
#pragma unroll
    for (int t = 0; t < 4; ++t) {
      float b0 = bf2[t * 16 + l15];
#pragma unroll
      for (int r = 0; r < 8; ++r) acc[t][r] = b0;
    }
    v16h a0 = load_a_lds(&Ah[w][l15 * 64], g);
    v16h a1 = load_a_lds(&Ah[w][l15 * 64 + 32], g);
#pragma unroll
    for (int t = 0; t < 4; ++t) {
      acc[t] = wmma16(a0, load_b(Wf2T, 64, t, 0, lane), acc[t]);
      acc[t] = wmma16(a1, load_b(Wf2T, 64, t, 1, lane), acc[t]);
    }
    int sv[8], dv[8];
    float cv[8];
#pragma unroll
    for (int r = 0; r < 8; ++r) {
      const int ed = g * 8 + r;
      sv[r] = Se[w][ed];
      dv[r] = De[w][ed];
      cv[r] = Ct[w][ed];
    }
#pragma unroll
    for (int t = 0; t < 4; ++t) {
      const int f = t * 16 + l15;
#pragma unroll
      for (int r = 0; r < 8; ++r) {
        const float wv = acc[t][r] * cv[r];
        const float xv = x1[(size_t)sv[r] * DF + f];
        atomicAdd(&agg[(size_t)dv[r] * DF + f], wv * xv);
      }
    }
  }
}

// ----------------------------------------------------- small helper kernels
__global__ void emb_gather_kernel(const int* __restrict__ z,
                                  const float* __restrict__ emb,
                                  float* __restrict__ feat, int n) {
  int idx = blockIdx.x * blockDim.x + threadIdx.x;
  if (idx < n * DF) {
    const int node = idx >> 6, d = idx & 63;
    feat[idx] = emb[(size_t)z[node] * DF + d];
  }
}

// Transpose+convert f32 W[K][Nc] -> f16 out[Nc][Kp] (zero pad k>=K).
__global__ void convertT_kernel(const float* __restrict__ Wsrc,
                                _Float16* __restrict__ out, int K, int Nc,
                                int Kp) {
  int idx = blockIdx.x * blockDim.x + threadIdx.x;
  if (idx < Nc * Kp) {
    const int nn = idx / Kp, k = idx % Kp;
    out[idx] = (_Float16)((k < K) ? Wsrc[k * Nc + nn] : 0.f);
  }
}

// h = feat - alpha*mean; feat <- h; accumulate sum(h^2) per feature.
__global__ void gn_pass1_kernel(float* __restrict__ feat,
                                const float* __restrict__ sumbuf,
                                const float* __restrict__ alpha,
                                float* __restrict__ sumsq, int n) {
  __shared__ float bsum[64];
  if (threadIdx.x < 64) bsum[threadIdx.x] = 0.f;
  __syncthreads();
  const int d = threadIdx.x & 63;  // blockDim multiple of 64
  const float mean = sumbuf[d] * (1.0f / (float)n);
  const float al = alpha[d];
  float local = 0.f;
  const int total = n * DF;
  for (int idx = blockIdx.x * blockDim.x + threadIdx.x; idx < total;
       idx += gridDim.x * blockDim.x) {
    const float h = feat[idx] - al * mean;
    feat[idx] = h;
    local += h * h;
  }
  atomicAdd(&bsum[d], local);
  __syncthreads();
  if (threadIdx.x < 64) atomicAdd(&sumsq[threadIdx.x], bsum[threadIdx.x]);
}

__global__ void gn_pass2_kernel(float* __restrict__ feat,
                                const float* __restrict__ sumsq,
                                const float* __restrict__ gamma,
                                const float* __restrict__ beta, int n) {
  const int d = threadIdx.x & 63;
  const float rs = rsqrtf(sumsq[d] * (1.0f / (float)n) + 1e-5f);
  const float gm = gamma[d] * rs, bt = beta[d];
  const int total = n * DF;
  for (int idx = blockIdx.x * blockDim.x + threadIdx.x; idx < total;
       idx += gridDim.x * blockDim.x)
    feat[idx] = feat[idx] * gm + bt;
}

__global__ void maxinit_kernel(float* __restrict__ maxb) {
  if (threadIdx.x < 64) maxb[threadIdx.x] = -__builtin_inff();
}

__global__ void r_max_kernel(const float* __restrict__ feat,
                             float* __restrict__ maxb, int n) {
  __shared__ float bmax[64];
  if (threadIdx.x < 64) bmax[threadIdx.x] = -__builtin_inff();
  __syncthreads();
  const int d = threadIdx.x & 63;
  float m = -__builtin_inff();
  const int total = n * DF;
  for (int idx = blockIdx.x * blockDim.x + threadIdx.x; idx < total;
       idx += gridDim.x * blockDim.x)
    m = fmaxf(m, feat[idx]);
  atomicMaxF(&bmax[d], m);
  __syncthreads();
  if (threadIdx.x < 64) atomicMaxF(&maxb[threadIdx.x], bmax[threadIdx.x]);
}

__global__ void r_sumexp_kernel(const float* __restrict__ feat,
                                const float* __restrict__ maxb,
                                float* __restrict__ s0,
                                float* __restrict__ s1, int n) {
  __shared__ float b0[64], b1[64];
  if (threadIdx.x < 64) { b0[threadIdx.x] = 0.f; b1[threadIdx.x] = 0.f; }
  __syncthreads();
  const int d = threadIdx.x & 63;
  const float mx = maxb[d];
  float a0 = 0.f, a1 = 0.f;
  const int total = n * DF;
  for (int idx = blockIdx.x * blockDim.x + threadIdx.x; idx < total;
       idx += gridDim.x * blockDim.x) {
    const float v = feat[idx];
    const float e = __expf(v - mx);
    a0 += e;
    a1 += e * v;
  }
  atomicAdd(&b0[d], a0);
  atomicAdd(&b1[d], a1);
  __syncthreads();
  if (threadIdx.x < 64) {
    atomicAdd(&s0[threadIdx.x], b0[threadIdx.x]);
    atomicAdd(&s1[threadIdx.x], b1[threadIdx.x]);
  }
}

__global__ void r_final_kernel(const float* __restrict__ s0,
                               const float* __restrict__ s1,
                               float* __restrict__ pooled) {
  if (threadIdx.x < 64) pooled[threadIdx.x] = s1[threadIdx.x] / s0[threadIdx.x];
}

// ------------------------------------------------------------------- launch
extern "C" void kernel_launch(void* const* d_in, const int* in_sizes, int n_in,
                              void* d_out, int out_size, void* d_ws,
                              size_t ws_size, hipStream_t stream) {
  (void)in_sizes; (void)n_in; (void)out_size; (void)ws_size;
  const int* z = (const int*)d_in[0];
  const int* src = (const int*)d_in[1];
  const int* dst = (const int*)d_in[2];
  const float* rij = (const float*)d_in[3];
  const float* emb = (const float*)d_in[4];
  const float* Win = (const float*)d_in[5];
  const float* Wf1 = (const float*)d_in[6];
  const float* bf1 = (const float*)d_in[7];
  const float* Wf2 = (const float*)d_in[8];
  const float* bf2 = (const float*)d_in[9];
  const float* Wo1 = (const float*)d_in[10];
  const float* bo1 = (const float*)d_in[11];
  const float* Wo2 = (const float*)d_in[12];
  const float* bo2 = (const float*)d_in[13];
  const float* ga = (const float*)d_in[14];
  const float* gg = (const float*)d_in[15];
  const float* gb = (const float*)d_in[16];

  float* feat = (float*)d_out;                     // [N,64]
  float* pooled = feat + (size_t)NNODES * DF;      // [64]

  const size_t nodeBytes = (size_t)NNODES * DF * sizeof(float);
  char* ws = (char*)d_ws;
  size_t off = 0;
  float* x1 = (float*)(ws + off);  off += nodeBytes;          // x1 / t1
  float* agg = (float*)(ws + off); off += nodeBytes;
  _Float16* WinT = (_Float16*)(ws + off); off += NLAYERS * 64 * 64 * 2;
  _Float16* Wf1T = (_Float16*)(ws + off); off += NLAYERS * 64 * 32 * 2;
  _Float16* Wf2T = (_Float16*)(ws + off); off += NLAYERS * 64 * 64 * 2;
  _Float16* Wo1T = (_Float16*)(ws + off); off += NLAYERS * 64 * 64 * 2;
  _Float16* Wo2T = (_Float16*)(ws + off); off += NLAYERS * 64 * 64 * 2;
  float* sumb = (float*)(ws + off);  off += 64 * sizeof(float);
  float* sumsq = (float*)(ws + off); off += 64 * sizeof(float);
  float* maxb = (float*)(ws + off);  off += 64 * sizeof(float);
  float* s0 = (float*)(ws + off);    off += 64 * sizeof(float);
  float* s1 = (float*)(ws + off);    off += 64 * sizeof(float);

  // Weight conversion (tiny; deterministic each call)
  for (int l = 0; l < NLAYERS; ++l) {
    convertT_kernel<<<16, 256, 0, stream>>>(Win + l * 64 * 64,
                                            WinT + l * 64 * 64, 64, 64, 64);
    convertT_kernel<<<8, 256, 0, stream>>>(Wf1 + l * NBASIS * 64,
                                           Wf1T + l * 64 * 32, NBASIS, 64, 32);
    convertT_kernel<<<16, 256, 0, stream>>>(Wf2 + l * 64 * 64,
                                            Wf2T + l * 64 * 64, 64, 64, 64);
    convertT_kernel<<<16, 256, 0, stream>>>(Wo1 + l * 64 * 64,
                                            Wo1T + l * 64 * 64, 64, 64, 64);
    convertT_kernel<<<16, 256, 0, stream>>>(Wo2 + l * 64 * 64,
                                            Wo2T + l * 64 * 64, 64, 64, 64);
  }

  // feat = emb[z]
  emb_gather_kernel<<<(NNODES * DF + 255) / 256, 256, 0, stream>>>(z, emb, feat,
                                                                   NNODES);

  const int nodeTiles = NNODES / 16;                    // 3125
  const int nodeGrid = (nodeTiles + 3) / 4;             // 782 (4 waves/block)
  const int edgeGrid = (NEDGES / 16 + 3) / 4;           // 25000

  for (int l = 0; l < NLAYERS; ++l) {
    hipMemsetAsync(agg, 0, nodeBytes, stream);
    hipMemsetAsync(sumb, 0, 64 * sizeof(float), stream);
    hipMemsetAsync(sumsq, 0, 64 * sizeof(float), stream);

    // x1 = feat @ Win[l]
    node_gemm_kernel<false, false, false, false><<<nodeGrid, 128, 0, stream>>>(
        feat, WinT + l * 64 * 64, nullptr, nullptr, x1, nullptr, NNODES);

    // edge message + scatter
    edge_kernel<<<edgeGrid, 128, 0, stream>>>(
        rij, src, dst, x1, Wf1T + l * 64 * 32, bf1 + l * 64,
        Wf2T + l * 64 * 64, bf2 + l * 64, agg, NEDGES);

    // t1 = swish(agg @ Wo1 + bo1)   (reuse x1 buffer)
    node_gemm_kernel<true, true, false, false><<<nodeGrid, 128, 0, stream>>>(
        agg, Wo1T + l * 64 * 64, bo1 + l * 64, nullptr, x1, nullptr, NNODES);

    // feat = feat + t1 @ Wo2 + bo2 ; accumulate per-feature sums
    node_gemm_kernel<true, false, true, true><<<nodeGrid, 128, 0, stream>>>(
        x1, Wo2T + l * 64 * 64, bo2 + l * 64, feat, feat, sumb, NNODES);

    // GraphNorm
    gn_pass1_kernel<<<1024, 256, 0, stream>>>(feat, sumb, ga + l * 64, sumsq,
                                              NNODES);
    gn_pass2_kernel<<<1024, 256, 0, stream>>>(feat, sumsq, gg + l * 64,
                                              gb + l * 64, NNODES);
  }

  // softmax readout
  maxinit_kernel<<<1, 64, 0, stream>>>(maxb);
  hipMemsetAsync(s0, 0, 64 * sizeof(float), stream);
  hipMemsetAsync(s1, 0, 64 * sizeof(float), stream);
  r_max_kernel<<<1024, 256, 0, stream>>>(feat, maxb, NNODES);
  r_sumexp_kernel<<<1024, 256, 0, stream>>>(feat, maxb, s0, s1, NNODES);
  r_final_kernel<<<1, 64, 0, stream>>>(s0, s1, pooled);
}
